// MultiHeadAttention_10505490006733
// MI455X (gfx1250) — matrix operand order repaired and checked by reading, not compiled, not run on hardware
//
#include <hip/hip_runtime.h>
#include <hip/hip_bf16.h>

typedef __bf16 bf16;
typedef __attribute__((ext_vector_type(16))) __bf16 v16bf;
typedef __attribute__((ext_vector_type(8)))  __bf16 v8bf;
typedef __attribute__((ext_vector_type(8)))  float  v8f;

#define HID 1024
#define NH 16
#define HD 64
#define BB 2
#define SS 2048
#define MTOT (BB*SS)   // 4096 rows

// ---- WMMA helpers (gfx1250, wave32) -------------------------------------

static __device__ __forceinline__ v16bf cat8(v8bf lo, v8bf hi) {
  return __builtin_shufflevector(lo, hi, 0,1,2,3,4,5,6,7,8,9,10,11,12,13,14,15);
}

// A-fragment (16x32 bf16): lane row = lane&15; K chunks
// {0..15} (lanes<16) / {16..31} (lanes>=16). Caller passes the
// per-lane pointer already offset by half*16 within the row.
static __device__ __forceinline__ v16bf loadA(const bf16* p) {
  v8bf lo = *(const v8bf*)(p);
  v8bf hi = *(const v8bf*)(p + 8);
  return cat8(lo, hi);
}

static __device__ __forceinline__ v8f wmma_bf16(v16bf a, v16bf b, v8f c) {
  // 8 args: (neg_a, A, neg_b, B, c_mod, C, reuse_a, reuse_b)
  return __builtin_amdgcn_wmma_f32_16x16x32_bf16(false, a, false, b, (short)0, c,
                                                 false, false);
}

static __device__ __forceinline__ v8f vzero8() {
  v8f z;
  #pragma unroll
  for (int i = 0; i < 8; i++) z[i] = 0.f;
  return z;
}

// ---- fp32 -> bf16 convert (x8 vectorized) --------------------------------

__global__ void cvt_f32_bf16_kernel(const float* __restrict__ src,
                                    bf16* __restrict__ dst, int n8) {
  int i = blockIdx.x * blockDim.x + threadIdx.x;
  if (i >= n8) return;
  const float4* s = (const float4*)src;
  float4 a = s[2*i], b = s[2*i + 1];
  v8bf o;
  o[0]=(bf16)a.x; o[1]=(bf16)a.y; o[2]=(bf16)a.z; o[3]=(bf16)a.w;
  o[4]=(bf16)b.x; o[5]=(bf16)b.y; o[6]=(bf16)b.z; o[7]=(bf16)b.w;
  *(v8bf*)(dst + 8*i) = o;
}

// ---- fused QKV projection: C = x @ W^T + bias ----------------------------
// 8 waves, 2x4 arrangement of 64x64 wave tiles -> 128x256 block tile.
// Per K-step: 4 A frags + 4 B frags (16 b128) feed 16 WMMA (reuse x4 each).
// blockIdx.z: 0=Q, 1=K, 2=V.  Q,K stored [b,h,s,hd]; V stored transposed
// [b,h,hd,s] so the P@V B-operand is a contiguous per-lane load.

__global__ __launch_bounds__(256) void qkv_gemm_kernel(
    const bf16* __restrict__ xb,
    const bf16* __restrict__ wqB, const bf16* __restrict__ wkB,
    const bf16* __restrict__ wvB,
    const float* __restrict__ bq, const float* __restrict__ bk,
    const float* __restrict__ bv,
    bf16* __restrict__ Qb, bf16* __restrict__ Kb, bf16* __restrict__ Vt) {
  const int lane = threadIdx.x & 31, half = lane >> 4, l16 = lane & 15;
  const int wave = threadIdx.x >> 5;
  const int wr0 = blockIdx.x * 128 + (wave & 1) * 64;   // 2 waves over rows
  const int wc0 = blockIdx.y * 256 + (wave >> 1) * 64;  // 4 waves over cols

  const bf16*  W    = (blockIdx.z == 0) ? wqB : (blockIdx.z == 1 ? wkB : wvB);
  const float* bias = (blockIdx.z == 0) ? bq  : (blockIdx.z == 1 ? bk  : bv);

  v8f acc[4][4];
  #pragma unroll
  for (int ri = 0; ri < 4; ri++)
    #pragma unroll
    for (int ci = 0; ci < 4; ci++) acc[ri][ci] = vzero8();

  const bf16* A = xb + (wr0 + l16) * HID + half * 16;
  const bf16* Bp = W + (wc0 + l16) * HID + half * 16;

  for (int k = 0; k < HID; k += 32) {
    v16bf a[4], b[4];
    #pragma unroll
    for (int ri = 0; ri < 4; ri++) a[ri] = loadA(A + ri * 16 * HID + k);
    #pragma unroll
    for (int ci = 0; ci < 4; ci++)
      b[ci] = *(const v16bf*)(Bp + ci * 16 * HID + k);
    #pragma unroll
    for (int ri = 0; ri < 4; ri++)
      #pragma unroll
      for (int ci = 0; ci < 4; ci++)
        acc[ri][ci] = wmma_bf16(a[ri], b[ci], acc[ri][ci]);
  }

  #pragma unroll
  for (int ri = 0; ri < 4; ri++)
  #pragma unroll
  for (int ci = 0; ci < 4; ci++) {
    const int n = wc0 + ci * 16 + l16;
    const float bb = bias[n];
    const int h = n >> 6, d = n & 63;
    #pragma unroll
    for (int e = 0; e < 8; e++) {
      const int m = wr0 + ri * 16 + e + 8 * half;   // C layout row
      const int b2 = m >> 11, s = m & 2047;
      const float v = acc[ri][ci][e] + bb;
      if (blockIdx.z < 2) {
        bf16* dst = (blockIdx.z == 0) ? Qb : Kb;
        dst[(((b2 * NH + h) * SS + s) << 6) + d] = (bf16)v;
      } else {
        Vt[(((b2 * NH + h) * HD + d) << 11) + s] = (bf16)v;
      }
    }
  }
}

// ---- causal flash attention ----------------------------------------------
// grid: (S/128 q-tiles, B*H). 4 waves/block; each wave owns 32 q-rows
// (2 row-tiles), so every K/V fragment load feeds two WMMAs. Per-wave causal
// trip count (no cross-wave dependencies; LDS regions are wave-private).

__global__ __launch_bounds__(128) void attn_kernel(
    const bf16* __restrict__ Qb, const bf16* __restrict__ Kb,
    const bf16* __restrict__ Vt, bf16* __restrict__ Ob) {
  __shared__ bf16 Pl[4 * 32 * 72];   // 72-elem padded rows: conflict-free b128
  const int lane = threadIdx.x & 31, half = lane >> 4, l16 = lane & 15;
  const int wave = threadIdx.x >> 5;
  const int bh = blockIdx.y;          // b*16 + h
  const int qr0 = blockIdx.x * 128 + wave * 32;

  // Q fragments for this wave's 32 rows, pre-scaled by 1/sqrt(64) (exact).
  v16bf aQ[2][2];
  #pragma unroll
  for (int ri = 0; ri < 2; ri++) {
    const bf16* Qrow = Qb + (bh * SS + qr0 + ri * 16 + l16) * HD;
    #pragma unroll
    for (int kk = 0; kk < 2; kk++) {
      v16bf a = loadA(Qrow + kk * 32 + half * 16);
      #pragma unroll
      for (int i = 0; i < 16; i++) a[i] = (bf16)((float)a[i] * 0.125f);
      aQ[ri][kk] = a;
    }
  }

  v8f oacc[2][4];
  float m_i[2][8], l_i[2][8];
  #pragma unroll
  for (int ri = 0; ri < 2; ri++) {
    #pragma unroll
    for (int dt = 0; dt < 4; dt++) oacc[ri][dt] = vzero8();
    #pragma unroll
    for (int r = 0; r < 8; r++) { m_i[ri][r] = -__builtin_inff(); l_i[ri][r] = 0.f; }
  }

  bf16* myP = Pl + wave * 32 * 72;
  const int nj = (qr0 >> 6) + 1;  // per-wave causal bound (64-key tiles)

  for (int j = 0; j < nj; j++) {
    // scores: 32 x 64 = 2 row-tiles x 4 col-tiles x 2 k-steps (16 WMMA)
    v8f sacc[2][4];
    #pragma unroll
    for (int ri = 0; ri < 2; ri++)
      #pragma unroll
      for (int t = 0; t < 4; t++) sacc[ri][t] = vzero8();
    #pragma unroll
    for (int t = 0; t < 4; t++) {
      const bf16* Krow = Kb + (bh * SS + j * 64 + t * 16 + l16) * HD + half * 16;
      v16bf b0 = *(const v16bf*)(Krow);
      v16bf b1 = *(const v16bf*)(Krow + 32);
      #pragma unroll
      for (int ri = 0; ri < 2; ri++) {
        sacc[ri][t] = wmma_bf16(aQ[ri][0], b0, sacc[ri][t]);
        sacc[ri][t] = wmma_bf16(aQ[ri][1], b1, sacc[ri][t]);
      }
    }

    if (j == nj - 1) {                // only the diagonal tile needs masking
      #pragma unroll
      for (int ri = 0; ri < 2; ri++)
      #pragma unroll
      for (int t = 0; t < 4; t++) {
        const int kc = j * 64 + t * 16 + l16;
        #pragma unroll
        for (int e = 0; e < 8; e++) {
          const int qq = qr0 + ri * 16 + e + 8 * half;
          if (kc > qq) sacc[ri][t][e] = -__builtin_inff();
        }
      }
    }

    // online softmax; row r of a half-wave lives across its 16 lanes
    #pragma unroll
    for (int ri = 0; ri < 2; ri++) {
      float alpha[8];
      #pragma unroll
      for (int r = 0; r < 8; r++) {
        float rm = fmaxf(fmaxf(sacc[ri][0][r], sacc[ri][1][r]),
                         fmaxf(sacc[ri][2][r], sacc[ri][3][r]));
        #pragma unroll
        for (int off = 8; off >= 1; off >>= 1)
          rm = fmaxf(rm, __shfl_xor(rm, off, 16));
        const float mn = fmaxf(m_i[ri][r], rm);
        const float al = __expf(m_i[ri][r] - mn);
        float rs = 0.f;
        #pragma unroll
        for (int t = 0; t < 4; t++) {
          const float p = __expf(sacc[ri][t][r] - mn);
          sacc[ri][t][r] = p;
          rs += p;
        }
        #pragma unroll
        for (int off = 8; off >= 1; off >>= 1) rs += __shfl_xor(rs, off, 16);
        l_i[ri][r] = l_i[ri][r] * al + rs;
        m_i[ri][r] = mn;
        alpha[r] = al;
      }
      #pragma unroll
      for (int dt = 0; dt < 4; dt++)
        #pragma unroll
        for (int r = 0; r < 8; r++) oacc[ri][dt][r] *= alpha[r];

      // P: C-layout -> LDS (wave-private region)
      #pragma unroll
      for (int t = 0; t < 4; t++)
        #pragma unroll
        for (int e = 0; e < 8; e++)
          myP[(ri * 16 + e + 8 * half) * 72 + t * 16 + l16] =
              (bf16)sacc[ri][t][e];
    }
    asm volatile("s_wait_dscnt 0" ::: "memory");

    v16bf aP[2][2];
    #pragma unroll
    for (int ri = 0; ri < 2; ri++)
      #pragma unroll
      for (int kk = 0; kk < 2; kk++) {
        const bf16* pr = myP + (ri * 16 + l16) * 72 + kk * 32 + half * 16;
        v8bf lo = *(const v8bf*)(pr);
        v8bf hi = *(const v8bf*)(pr + 8);
        aP[ri][kk] = cat8(lo, hi);
      }

    // O += P @ V : V stored transposed [d, key]; each V frag feeds 2 WMMA
    #pragma unroll
    for (int dt = 0; dt < 4; dt++) {
      const bf16* Vrow =
          Vt + (bh * HD + dt * 16 + l16) * SS + j * 64 + half * 16;
      v16bf b0 = *(const v16bf*)(Vrow);
      v16bf b1 = *(const v16bf*)(Vrow + 32);
      #pragma unroll
      for (int ri = 0; ri < 2; ri++) {
        oacc[ri][dt] = wmma_bf16(aP[ri][0], b0, oacc[ri][dt]);
        oacc[ri][dt] = wmma_bf16(aP[ri][1], b1, oacc[ri][dt]);
      }
    }
  }

  // epilogue: O /= l, store [b, s, h*64+d] bf16 for the output projection
  const int b = bh >> 4, h = bh & 15;
  #pragma unroll
  for (int ri = 0; ri < 2; ri++)
  #pragma unroll
  for (int r = 0; r < 8; r++) {
    const float inv = 1.f / l_i[ri][r];
    const int qq = qr0 + ri * 16 + r + 8 * half;
    #pragma unroll
    for (int dt = 0; dt < 4; dt++)
      Ob[(b * SS + qq) * HID + h * HD + dt * 16 + l16] =
          (bf16)(oacc[ri][dt][r] * inv);
  }
}

// ---- output projection: out = O @ Wo^T + bo (fp32 out) -------------------

__global__ __launch_bounds__(256) void out_gemm_kernel(
    const bf16* __restrict__ Ab, const bf16* __restrict__ Wb,
    const float* __restrict__ bo, float* __restrict__ out) {
  const int lane = threadIdx.x & 31, half = lane >> 4, l16 = lane & 15;
  const int wave = threadIdx.x >> 5;
  const int wr0 = blockIdx.x * 128 + (wave & 1) * 64;
  const int wc0 = blockIdx.y * 256 + (wave >> 1) * 64;

  v8f acc[4][4];
  #pragma unroll
  for (int ri = 0; ri < 4; ri++)
    #pragma unroll
    for (int ci = 0; ci < 4; ci++) acc[ri][ci] = vzero8();

  const bf16* A = Ab + (wr0 + l16) * HID + half * 16;
  const bf16* Bp = Wb + (wc0 + l16) * HID + half * 16;

  for (int k = 0; k < HID; k += 32) {
    v16bf a[4], b[4];
    #pragma unroll
    for (int ri = 0; ri < 4; ri++) a[ri] = loadA(A + ri * 16 * HID + k);
    #pragma unroll
    for (int ci = 0; ci < 4; ci++)
      b[ci] = *(const v16bf*)(Bp + ci * 16 * HID + k);
    #pragma unroll
    for (int ri = 0; ri < 4; ri++)
      #pragma unroll
      for (int ci = 0; ci < 4; ci++)
        acc[ri][ci] = wmma_bf16(a[ri], b[ci], acc[ri][ci]);
  }

  #pragma unroll
  for (int ri = 0; ri < 4; ri++)
  #pragma unroll
  for (int ci = 0; ci < 4; ci++) {
    const int n = wc0 + ci * 16 + l16;
    const float bb = bo[n];
    #pragma unroll
    for (int e = 0; e < 8; e++) {
      const int m = wr0 + ri * 16 + e + 8 * half;
      out[m * HID + n] = acc[ri][ci][e] + bb;
    }
  }
}

// ---- launch ---------------------------------------------------------------

extern "C" void kernel_launch(void* const* d_in, const int* in_sizes, int n_in,
                              void* d_out, int out_size, void* d_ws,
                              size_t ws_size, hipStream_t stream) {
  (void)in_sizes; (void)n_in; (void)out_size; (void)ws_size;
  const float* x  = (const float*)d_in[0];
  const float* Wq = (const float*)d_in[1];
  const float* bq = (const float*)d_in[2];
  const float* Wk = (const float*)d_in[3];
  const float* bk = (const float*)d_in[4];
  const float* Wv = (const float*)d_in[5];
  const float* bv = (const float*)d_in[6];
  const float* Wo = (const float*)d_in[7];
  const float* bo = (const float*)d_in[8];

  char* ws = (char*)d_ws;
  const size_t MiB = (size_t)1 << 20;
  bf16* xb  = (bf16*)(ws +  0 * MiB);   // [4096,1024]        8 MiB
  bf16* wqB = (bf16*)(ws +  8 * MiB);   // [1024,1024]        2 MiB
  bf16* wkB = (bf16*)(ws + 10 * MiB);
  bf16* wvB = (bf16*)(ws + 12 * MiB);
  bf16* woB = (bf16*)(ws + 14 * MiB);
  bf16* Qb  = (bf16*)(ws + 16 * MiB);   // [b,h,s,hd]         8 MiB
  bf16* Kb  = (bf16*)(ws + 24 * MiB);   // [b,h,s,hd]         8 MiB
  bf16* Vt  = (bf16*)(ws + 32 * MiB);   // [b,h,hd,s]         8 MiB
  bf16* Ob  = (bf16*)(ws + 40 * MiB);   // [b,s,h*hd]         8 MiB

  const int nx8 = MTOT * HID / 8;       // 524288
  const int nw8 = HID * HID / 8;        // 131072
  cvt_f32_bf16_kernel<<<(nx8 + 255) / 256, 256, 0, stream>>>(x,  xb,  nx8);
  cvt_f32_bf16_kernel<<<(nw8 + 255) / 256, 256, 0, stream>>>(Wq, wqB, nw8);
  cvt_f32_bf16_kernel<<<(nw8 + 255) / 256, 256, 0, stream>>>(Wk, wkB, nw8);
  cvt_f32_bf16_kernel<<<(nw8 + 255) / 256, 256, 0, stream>>>(Wv, wvB, nw8);
  cvt_f32_bf16_kernel<<<(nw8 + 255) / 256, 256, 0, stream>>>(Wo, woB, nw8);

  qkv_gemm_kernel<<<dim3(MTOT / 128, HID / 256, 3), 256, 0, stream>>>(
      xb, wqB, wkB, wvB, bq, bk, bv, Qb, Kb, Vt);

  attn_kernel<<<dim3(SS / 128, BB * NH), 128, 0, stream>>>(Qb, Kb, Vt, Ob);

  out_gemm_kernel<<<dim3(MTOT / 128, HID / 256), 256, 0, stream>>>(
      Ob, woB, bo, (float*)d_out);
}
